// LSTM_24910810317051
// MI455X (gfx1250) — matrix operand order repaired and checked by reading, not compiled, not run on hardware
//
#include <hip/hip_runtime.h>

// ---------------- types ----------------
typedef __bf16 bf16_t;
typedef __attribute__((ext_vector_type(16))) __bf16 v16bf;
typedef __attribute__((ext_vector_type(8)))  __bf16 v8bf;
typedef __attribute__((ext_vector_type(8)))  float  v8f;

// Problem constants (from reference: N=64, T=512, D=1024, H=1024)
constexpr int kN = 64;
constexpr int kT = 512;
constexpr int kD = 1024;
constexpr int kH = 1024;
constexpr int kG = 4 * kH;   // 4096 gate columns

// Persistent grid: 8 blocks x 8 waves = 64 waves (one per 16-wide j-tile).
// Each wave owns ALL 4 batch tiles (M=64) for its 16 hidden units.
constexpr int kBlocks  = 8;
constexpr int kThreads = 256;

// LDS staging: rows padded by 8 bf16 (16B) so ds_load_b128 A-fragment reads
// spread across all 64 LDS banks (row stride 2064 B -> lane r hits banks 4r..4r+3).
constexpr int kPad  = 8;
constexpr int kLdsW = kH + kPad;          // padded row length (elements)

// ---------------- workspace layout (bytes) ----------------
constexpr size_t OFF_X   = 0;
constexpr size_t OFF_WXT = OFF_X   + (size_t)kN * kT * kD * 2;
constexpr size_t OFF_WHT = OFF_WXT + (size_t)kG * kD * 2;
constexpr size_t OFF_H   = OFF_WHT + (size_t)kG * kH * 2;
constexpr size_t OFF_C   = OFF_H   + (size_t)2 * kN * kH * 2;
constexpr size_t OFF_BAR = OFF_C   + (size_t)kN * kH * 4;
constexpr size_t OFF_XW  = OFF_BAR + 256;                       // optional big buffer
constexpr size_t WS_NEED_XW = OFF_XW + (size_t)kT * kN * kG * 4; // ~620 MB total

// ---------------- helpers ----------------
static __device__ __forceinline__ v8f vzero8() {
    v8f z;
#pragma unroll
    for (int i = 0; i < 8; ++i) z[i] = 0.0f;
    return z;
}
static __device__ __forceinline__ float sigmoid_f(float x) {
    return 1.0f / (1.0f + __expf(-x));
}
static __device__ __forceinline__ float tanh_f(float x) {
    return 1.0f - 2.0f / (__expf(2.0f * x) + 1.0f);   // saturates to +-1, no NaN
}

// Device-scope grid barrier (persistent kernel). CDNA5: stores land in L2
// (WGP$ read-only) -> release waits own stores; acquire invalidates WGP$.
static __device__ __forceinline__ void grid_sync(unsigned* cnt, unsigned target) {
    __threadfence();
    __syncthreads();
    if (threadIdx.x == 0) {
        __hip_atomic_fetch_add(cnt, 1u, __ATOMIC_RELEASE, __HIP_MEMORY_SCOPE_AGENT);
        while (__hip_atomic_load(cnt, __ATOMIC_ACQUIRE, __HIP_MEMORY_SCOPE_AGENT) < target) {
            __builtin_amdgcn_s_sleep(2);
        }
    }
    __syncthreads();
    __builtin_amdgcn_fence(__ATOMIC_ACQUIRE, "agent");
}

// ---------------- prep kernels ----------------
__global__ void cvt_bf16_kernel(const float* __restrict__ in,
                                bf16_t* __restrict__ out, size_t n) {
    size_t i = (size_t)blockIdx.x * blockDim.x + threadIdx.x;
    if (i < n) out[i] = (bf16_t)in[i];
}

// W: K x C row-major fp32 -> WT: C x K row-major bf16
__global__ void transpose_bf16_kernel(const float* __restrict__ W,
                                      bf16_t* __restrict__ WT, int K, int C) {
    size_t i = (size_t)blockIdx.x * blockDim.x + threadIdx.x;
    if (i < (size_t)K * C) {
        int k = (int)(i / C);
        int j = (int)(i % C);
        WT[(size_t)j * K + k] = (bf16_t)W[i];
    }
}

__global__ void init_state_kernel(const float* __restrict__ h0,
                                  bf16_t* __restrict__ hbuf0,
                                  float* __restrict__ c_ws,
                                  unsigned* __restrict__ bar, int n) {
    int i = blockIdx.x * blockDim.x + threadIdx.x;
    if (i < n) {
        hbuf0[i] = (bf16_t)h0[i];
        c_ws[i]  = 0.0f;
    }
    if (i == 0) *bar = 0u;
}

// ---------------- phase 1: full-occupancy xW GEMM ----------------
// xw[t][n][g] = x[n][t][:] . Wx[:][g]   (bf16 WMMA, fp32 out)
// One wave per 16x16 output tile of the (T*N=32768) x 4096 matrix.
__global__ __launch_bounds__(kThreads)
void xw_gemm_kernel(const bf16_t* __restrict__ xbf,  // [N][T][D]
                    const bf16_t* __restrict__ wxT,  // [4H][D]
                    float* __restrict__ xw) {        // [T][N][4H]
    const int lane  = threadIdx.x & 31;
    const int wave  = blockIdx.x * (blockDim.x >> 5) + (threadIdx.x >> 5);
    const int mt    = wave >> 8;          // 0..2047  (row tile over T*N)
    const int nt    = wave & 255;         // 0..255   (col tile over 4H)
    const int lhalf = lane >> 4;
    const int l16   = lane & 15;
    const int akb   = lhalf * 16;
    const int bkb   = lhalf * 16;

    const int r = mt * 16 + l16;          // output row = t*64 + n
    const int t = r >> 6;
    const int n = r & 63;

    const bf16_t* arow = xbf + ((size_t)n * kT + t) * kD + akb;
    const bf16_t* brow = wxT + (size_t)(nt * 16 + l16) * kD + bkb;

    union AU { v16bf v; v8bf h[2]; };
    v8f acc = vzero8();
#pragma unroll 2
    for (int k = 0; k < kD; k += 32) {
        AU a;
        a.h[0] = *(const v8bf*)(arow + k);
        a.h[1] = *(const v8bf*)(arow + k + 8);
        v16bf b = *(const v16bf*)(brow + k);
        acc = __builtin_amdgcn_wmma_f32_16x16x32_bf16(
            false, a.v, false, b, (short)0, acc, false, false);
    }
    // C/D layout: VGPR v, lanes 0-15 -> M=v, N=lane; lanes 16-31 -> M=v+8.
#pragma unroll
    for (int v = 0; v < 8; ++v) {
        const size_t row = (size_t)mt * 16 + v + lhalf * 8;
        xw[row * kG + nt * 16 + l16] = acc[v];
    }
}

// ---------------- persistent fused LSTM recurrence ----------------
__global__ __launch_bounds__(kThreads)
void lstm_persistent_kernel(const bf16_t* __restrict__ xbf,   // [N][T][D]
                            const bf16_t* __restrict__ wxT,   // [4H][D]
                            const bf16_t* __restrict__ whT,   // [4H][H]
                            const float*  __restrict__ xw,    // [T][N][4H] or null
                            const float*  __restrict__ bias,  // [4H]
                            float*        __restrict__ c_ws,  // [N][H]
                            bf16_t*       __restrict__ hbuf0, // [N][H]
                            bf16_t*       __restrict__ hbuf1, // [N][H]
                            float*        __restrict__ out,   // [N][T][H]
                            unsigned*     __restrict__ bar,
                            int use_xw) {
    // 264 KB static LDS: h slab (always) + x slab (fused fallback path)
    __shared__ __align__(16) bf16_t sH[kN * kLdsW];
    __shared__ __align__(16) bf16_t sX[kN * kLdsW];

    const int lane  = threadIdx.x & 31;
    const int wave  = blockIdx.x * (blockDim.x >> 5) + (threadIdx.x >> 5);
    const int jtile = wave;        // 0..63
    const int j0    = jtile * 16;
    const int lhalf = lane >> 4;
    const int l16   = lane & 15;
    const int akb   = lhalf * 16;  // A frag: K {akb..akb+15}
    const int bkb   = lhalf * 16;  // B frag: K {bkb..bkb+15}

    // t-invariant per-lane pointers
    const bf16_t* sHrow[4];
    const bf16_t* sXrow[4];
#pragma unroll
    for (int mt = 0; mt < 4; ++mt) {
        const int arow = mt * 16 + l16;
        sHrow[mt] = sH + (size_t)arow * kLdsW + akb;
        sXrow[mt] = sX + (size_t)arow * kLdsW + akb;
    }
    const bf16_t* bx[4];
    const bf16_t* bh[4];
#pragma unroll
    for (int g = 0; g < 4; ++g) {
        bx[g] = wxT + (size_t)(g * kH + j0 + l16) * kD + bkb;
        bh[g] = whT + (size_t)(g * kH + j0 + l16) * kH + bkb;
    }

    const int col = j0 + l16;
    const float bi  = bias[0 * kH + col];
    const float bf_ = bias[1 * kH + col];
    const float bo  = bias[2 * kH + col];
    const float bg  = bias[3 * kH + col];

    union AU { v16bf v; v8bf h[2]; };

    for (int t = 0; t < kT; ++t) {
        // ---- stage A slabs into LDS (coalesced global -> ds_store_b128) ----
        const bf16_t* hsrc = (t & 1) ? hbuf1 : hbuf0;
        for (int i = threadIdx.x; i < kN * (kH / 8); i += kThreads) {
            const int row = i >> 7;           // kH/8 = 128 chunks per row
            const int c8  = (i & 127) * 8;
            *(v8bf*)(sH + (size_t)row * kLdsW + c8) =
                *(const v8bf*)(hsrc + (size_t)row * kH + c8);
        }
        if (!use_xw) {
            for (int i = threadIdx.x; i < kN * (kD / 8); i += kThreads) {
                const int row = i >> 7;
                const int c8  = (i & 127) * 8;
                *(v8bf*)(sX + (size_t)row * kLdsW + c8) =
                    *(const v8bf*)(xbf + ((size_t)row * kT + t) * kD + c8);
            }
        }
        __syncthreads();

        // ---- init accumulators: precomputed xW[t] or zero ----
        v8f acc[4][4];   // [gate][mtile]
        if (use_xw) {
            const float* xwt = xw + (size_t)t * kN * kG;
#pragma unroll
            for (int g = 0; g < 4; ++g)
#pragma unroll
                for (int mt = 0; mt < 4; ++mt)
#pragma unroll
                    for (int r = 0; r < 8; ++r) {
                        const int m = mt * 16 + r + lhalf * 8;
                        acc[g][mt][r] = xwt[(size_t)m * kG + g * kH + col];
                    }
        } else {
#pragma unroll
            for (int g = 0; g < 4; ++g)
#pragma unroll
                for (int mt = 0; mt < 4; ++mt) acc[g][mt] = vzero8();

            // fused x_t @ Wx (fallback when xW buffer doesn't fit in ws)
#pragma unroll 1
            for (int k = 0; k < kD; k += 32) {
                AU a[4];
#pragma unroll
                for (int mt = 0; mt < 4; ++mt) {
                    a[mt].h[0] = *(const v8bf*)(sXrow[mt] + k);
                    a[mt].h[1] = *(const v8bf*)(sXrow[mt] + k + 8);
                }
#pragma unroll
                for (int g = 0; g < 4; ++g) {
                    v16bf bfrag = *(const v16bf*)(bx[g] + k);
#pragma unroll
                    for (int mt = 0; mt < 4; ++mt)
                        acc[g][mt] = __builtin_amdgcn_wmma_f32_16x16x32_bf16(
                            false, a[mt].v, false, bfrag, (short)0, acc[g][mt],
                            false, false);
                }
            }
        }

        // ---- h_{t-1} @ Wh : A from LDS, B rows wave-exclusive from L2 ----
#pragma unroll 1
        for (int k = 0; k < kH; k += 32) {
            AU a[4];
#pragma unroll
            for (int mt = 0; mt < 4; ++mt) {
                a[mt].h[0] = *(const v8bf*)(sHrow[mt] + k);
                a[mt].h[1] = *(const v8bf*)(sHrow[mt] + k + 8);
            }
#pragma unroll
            for (int g = 0; g < 4; ++g) {
                v16bf bfrag = *(const v16bf*)(bh[g] + k);
#pragma unroll
                for (int mt = 0; mt < 4; ++mt)
                    acc[g][mt] = __builtin_amdgcn_wmma_f32_16x16x32_bf16(
                        false, a[mt].v, false, bfrag, (short)0, acc[g][mt],
                        false, false);
            }
        }

        // ---- gates + cell/hidden update ----
        bf16_t* hnext = (t & 1) ? hbuf0 : hbuf1;
#pragma unroll
        for (int mt = 0; mt < 4; ++mt) {
#pragma unroll
            for (int r = 0; r < 8; ++r) {
                const int m = mt * 16 + r + lhalf * 8;
                const float iv = sigmoid_f(acc[0][mt][r] + bi);
                const float fv = sigmoid_f(acc[1][mt][r] + bf_);
                const float ov = sigmoid_f(acc[2][mt][r] + bo);
                const float gv = tanh_f(acc[3][mt][r] + bg);

                const size_t sidx = (size_t)m * kH + col;
                const float c_old = c_ws[sidx];
                const float c_new = fv * c_old + iv * gv;
                c_ws[sidx] = c_new;

                const float hv = ov * tanh_f(c_new);
                out[((size_t)m * kT + t) * kH + col] = hv;
                hnext[sidx] = (bf16_t)hv;
            }
        }

        grid_sync(bar, (unsigned)(gridDim.x * (t + 1)));
    }
}

// ---------------- host entry ----------------
extern "C" void kernel_launch(void* const* d_in, const int* in_sizes, int n_in,
                              void* d_out, int out_size, void* d_ws, size_t ws_size,
                              hipStream_t stream) {
    const float* x  = (const float*)d_in[0];  // (N,T,D)
    const float* h0 = (const float*)d_in[1];  // (N,H)
    const float* Wx = (const float*)d_in[2];  // (D,4H)
    const float* Wh = (const float*)d_in[3];  // (H,4H)
    const float* b  = (const float*)d_in[4];  // (4H,)
    float* out = (float*)d_out;               // (N,T,H)

    char* ws = (char*)d_ws;
    bf16_t*   xbf  = (bf16_t*)(ws + OFF_X);
    bf16_t*   wxT  = (bf16_t*)(ws + OFF_WXT);
    bf16_t*   whT  = (bf16_t*)(ws + OFF_WHT);
    bf16_t*   hbuf = (bf16_t*)(ws + OFF_H);
    float*    c_ws = (float*) (ws + OFF_C);
    unsigned* bar  = (unsigned*)(ws + OFF_BAR);
    float*    xw   = (float*)  (ws + OFF_XW);

    const int use_xw = (ws_size >= WS_NEED_XW) ? 1 : 0;

    // ---- prep ----
    const size_t nx = (size_t)kN * kT * kD;
    cvt_bf16_kernel<<<(unsigned)((nx + 255) / 256), 256, 0, stream>>>(x, xbf, nx);
    const size_t nw = (size_t)kD * kG;
    transpose_bf16_kernel<<<(unsigned)((nw + 255) / 256), 256, 0, stream>>>(Wx, wxT, kD, kG);
    transpose_bf16_kernel<<<(unsigned)((nw + 255) / 256), 256, 0, stream>>>(Wh, whT, kH, kG);
    const int ns = kN * kH;
    init_state_kernel<<<(ns + 255) / 256, 256, 0, stream>>>(h0, hbuf, c_ws, bar, ns);

    // ---- phase 1: full-occupancy input projection (if ws fits) ----
    if (use_xw) {
        // (T*N/16) * (4H/16) tiles = 524288 waves / 8 per block
        xw_gemm_kernel<<<65536, kThreads, 0, stream>>>(xbf, wxT, xw);
    }

    // ---- phase 2: ONE persistent launch runs all 512 steps ----
    const size_t NH = (size_t)kN * kH;
    lstm_persistent_kernel<<<kBlocks, kThreads, 0, stream>>>(
        xbf, wxT, whT, use_xw ? xw : nullptr, b, c_ws,
        hbuf, hbuf + NH, out, bar, use_xw);
}
